// MHAtt_25795573580579
// MI455X (gfx1250) — matrix operand order repaired and checked by reading, not compiled, not run on hardware
//
#include <hip/hip_runtime.h>
#include <hip/hip_bf16.h>
#include <stdint.h>

// MHA pipeline for gfx1250 (MI455X): all GEMMs via v_wmma_f32_16x16x32_f16,
// double-buffered LDS staging, async global->LDS copies where contiguous.
// B=16, S=1024, RD=NH=1024, H=8, dk=128.

typedef _Float16 v16h __attribute__((ext_vector_type(16)));
typedef _Float16 v8h  __attribute__((ext_vector_type(8)));
typedef _Float16 v4h  __attribute__((ext_vector_type(4)));
typedef float    v8f  __attribute__((ext_vector_type(8)));
typedef float    v4f  __attribute__((ext_vector_type(4)));
typedef int      v4i  __attribute__((ext_vector_type(4)));

#define AS_LD 40   // 32 data halves + 8 pad (80B rows, keeps 16B alignment)
#define BS_LD 40

#if __has_builtin(__builtin_amdgcn_global_load_async_to_lds_b128)
#define HAVE_ASYNC 1
#else
#define HAVE_ASYNC 0
#endif

// 16-byte async global->LDS copy (one GLOBAL_LOAD_ASYNC_TO_LDS_B128 per lane).
__device__ __forceinline__ void async_copy16(const _Float16* g, _Float16* l) {
#if HAVE_ASYNC
    __builtin_amdgcn_global_load_async_to_lds_b128(
        (__attribute__((address_space(1))) v4i*)g,
        (__attribute__((address_space(3))) v4i*)l,
        0, 0);
#endif
}

__device__ __forceinline__ void wait_async_zero() {
#if HAVE_ASYNC
#if __has_builtin(__builtin_amdgcn_s_wait_asynccnt)
    __builtin_amdgcn_s_wait_asynccnt(0);
#else
    asm volatile("s_wait_asynccnt 0x0" ::: "memory");
#endif
#endif
}

// ---------------------------------------------------------------------------
// f32 -> f16 elementwise convert, 4 elements/thread
// ---------------------------------------------------------------------------
__global__ __launch_bounds__(256) void cvt_f32_f16_x4(const float* __restrict__ in,
                                                      _Float16* __restrict__ out, int n4) {
    int i = blockIdx.x * 256 + threadIdx.x;
    if (i < n4) {
        v4f x = ((const v4f*)in)[i];
        v4h y;
        y[0] = (_Float16)x[0]; y[1] = (_Float16)x[1];
        y[2] = (_Float16)x[2]; y[3] = (_Float16)x[3];
        ((v4h*)out)[i] = y;
    }
}

// ---------------------------------------------------------------------------
// Batched WMMA GEMM, all-f16 operands, f32 accumulate.
//   Block: 256 threads = 8 waves; block tile 128(M) x 128(N); k-step 32.
//   Waves: (wm,wn) = (wv>>1, wv&1); wave tile 32(M) x 64(N) -> 2x4 accs.
//   B_IS_NT : B source is [N,Kd] row-major (B^T, contiguous in kd -> async ok)
//             else [Kd,N] row-major (transpose-scatter through VGPRs).
//   LAYOUT  : 0 = C[z*c_bstride + row*ldc + col]
//             1 = split heads:  z=b, C[b][col>>7][row][col&127]   (S=1024,dk=128)
//             2 = merge heads:  z=b*8+h, C[b][row][h*128+col]     (NH=1024)
// ---------------------------------------------------------------------------
template<bool B_IS_NT, int LAYOUT, typename OT>
__global__ __launch_bounds__(256) void gemm_wmma(
    int Kd,
    const _Float16* __restrict__ A, int64_t a_bstride, int lda,
    const _Float16* __restrict__ Bm, int64_t b_bstride, int ldb,
    const float* __restrict__ bias, float scale,
    OT* __restrict__ Cp, int64_t c_bstride, int ldc)
{
    __shared__ _Float16 As[2][128][AS_LD];    // [buf][m][kd]
    __shared__ _Float16 BsT[2][128][BS_LD];   // [buf][n][kd]

    const int t    = threadIdx.x;
    const int lane = t & 31;
    const int wv   = t >> 5;
    const int lrow = lane & 15;
    const int lhi  = lane >> 4;
    const int wm   = wv >> 1;      // 0..3
    const int wn   = wv & 1;       // 0..1
    const int bm0  = blockIdx.x * 128;
    const int bn0  = blockIdx.y * 128;
    const int z    = blockIdx.z;

    const _Float16* Ab = A  + (int64_t)z * a_bstride;
    const _Float16* Bb = Bm + (int64_t)z * b_bstride;

    // contiguous 16B-chunk mapping (A tile and NT-B tile are 128 rows x 32 halves)
    const int cr0 = t >> 2;            // rows 0..63
    const int cc  = (t & 3) * 8;       // half-cols {0,8,16,24}
    const int cr1 = cr0 + 64;          // rows 64..127
    // normal-B (transpose) mapping: 32(kd) x 128(n), 16 halves per thread
    const int bkd = t >> 3;            // 0..31
    const int bnb = (t & 7) * 16;      // 0..112

    v8f acc[2][4] = {};
    v8h ra0, ra1;                      // reg staging (fallback A / normal B)
    v8h rb0, rb1;

    const int nsteps = Kd / 32;
    int buf = 0;

    // ---- prologue: fetch step 0 ----
#if HAVE_ASYNC
    async_copy16(Ab + (int64_t)(bm0 + cr0) * lda + cc, &As[0][cr0][cc]);
    async_copy16(Ab + (int64_t)(bm0 + cr1) * lda + cc, &As[0][cr1][cc]);
    if constexpr (B_IS_NT) {
        async_copy16(Bb + (int64_t)(bn0 + cr0) * ldb + cc, &BsT[0][cr0][cc]);
        async_copy16(Bb + (int64_t)(bn0 + cr1) * ldb + cc, &BsT[0][cr1][cc]);
    }
#else
    ra0 = *(const v8h*)(Ab + (int64_t)(bm0 + cr0) * lda + cc);
    ra1 = *(const v8h*)(Ab + (int64_t)(bm0 + cr1) * lda + cc);
    if constexpr (B_IS_NT) {
        rb0 = *(const v8h*)(Bb + (int64_t)(bn0 + cr0) * ldb + cc);
        rb1 = *(const v8h*)(Bb + (int64_t)(bn0 + cr1) * ldb + cc);
    }
#endif
    if constexpr (!B_IS_NT) {
        rb0 = *(const v8h*)(Bb + (int64_t)bkd * ldb + bn0 + bnb);
        rb1 = *(const v8h*)(Bb + (int64_t)bkd * ldb + bn0 + bnb + 8);
    }

    for (int i = 0; i < nsteps; ++i) {
        const bool more = (i + 1 < nsteps);
        const int ksn = (i + 1) * 32;

        // ---- commit reg-staged data for step i into LDS[buf] ----
#if !HAVE_ASYNC
        *(v8h*)&As[buf][cr0][cc] = ra0;
        *(v8h*)&As[buf][cr1][cc] = ra1;
        if constexpr (B_IS_NT) {
            *(v8h*)&BsT[buf][cr0][cc] = rb0;
            *(v8h*)&BsT[buf][cr1][cc] = rb1;
        }
#endif
        if constexpr (!B_IS_NT) {
            #pragma unroll
            for (int j = 0; j < 8; j++) {
                BsT[buf][bnb + j][bkd]     = rb0[j];
                BsT[buf][bnb + 8 + j][bkd] = rb1[j];
            }
        }

        // ---- reg fetches for step i+1 (fly during wait/compute) ----
        if (more) {
#if !HAVE_ASYNC
            ra0 = *(const v8h*)(Ab + (int64_t)(bm0 + cr0) * lda + ksn + cc);
            ra1 = *(const v8h*)(Ab + (int64_t)(bm0 + cr1) * lda + ksn + cc);
            if constexpr (B_IS_NT) {
                rb0 = *(const v8h*)(Bb + (int64_t)(bn0 + cr0) * ldb + ksn + cc);
                rb1 = *(const v8h*)(Bb + (int64_t)(bn0 + cr1) * ldb + ksn + cc);
            }
#endif
            if constexpr (!B_IS_NT) {
                rb0 = *(const v8h*)(Bb + (int64_t)(ksn + bkd) * ldb + bn0 + bnb);
                rb1 = *(const v8h*)(Bb + (int64_t)(ksn + bkd) * ldb + bn0 + bnb + 8);
            }
        }

        // ---- make step-i staging complete & visible; all waves past compute(i-1) ----
        wait_async_zero();
        __syncthreads();

        // ---- async fetches for step i+1 into LDS[buf^1] (safe: all waves are
        //      done reading buf^1, which was compute(i-1)'s buffer) ----
#if HAVE_ASYNC
        if (more) {
            async_copy16(Ab + (int64_t)(bm0 + cr0) * lda + ksn + cc, &As[buf ^ 1][cr0][cc]);
            async_copy16(Ab + (int64_t)(bm0 + cr1) * lda + ksn + cc, &As[buf ^ 1][cr1][cc]);
            if constexpr (B_IS_NT) {
                async_copy16(Bb + (int64_t)(bn0 + cr0) * ldb + ksn + cc, &BsT[buf ^ 1][cr0][cc]);
                async_copy16(Bb + (int64_t)(bn0 + cr1) * ldb + ksn + cc, &BsT[buf ^ 1][cr1][cc]);
            }
        }
#endif

        // ---- load all fragments, then 8 back-to-back WMMAs ----
        v16h af[2];
        #pragma unroll
        for (int ms = 0; ms < 2; ms++) {
            const int r = wm * 32 + ms * 16 + lrow;
            v8h lo = *(const v8h*)&As[buf][r][lhi * 8];
            v8h hi = *(const v8h*)&As[buf][r][16 + lhi * 8];
            af[ms] = __builtin_shufflevector(lo, hi,
                0,1,2,3,4,5,6,7,8,9,10,11,12,13,14,15);
        }
        v16h bf[4];
        #pragma unroll
        for (int ns = 0; ns < 4; ns++) {
            const int n = wn * 64 + ns * 16 + lrow;
            v8h lo = *(const v8h*)&BsT[buf][n][lhi * 8];
            v8h hi = *(const v8h*)&BsT[buf][n][16 + lhi * 8];
            bf[ns] = __builtin_shufflevector(lo, hi,
                0,1,2,3,4,5,6,7,8,9,10,11,12,13,14,15);
        }
        #pragma unroll
        for (int ms = 0; ms < 2; ms++)
            #pragma unroll
            for (int ns = 0; ns < 4; ns++)
                acc[ms][ns] = __builtin_amdgcn_wmma_f32_16x16x32_f16(
                    false, af[ms], false, bf[ns], (short)0, acc[ms][ns], false, false);

        buf ^= 1;
    }

    // ---- epilogue: C/D layout M = r + 8*lhi, N = lrow ----
    #pragma unroll
    for (int ms = 0; ms < 2; ms++) {
        #pragma unroll
        for (int ns = 0; ns < 4; ns++) {
            const int col = bn0 + wn * 64 + ns * 16 + lrow;
            const float bv = bias ? bias[col] : 0.0f;
            #pragma unroll
            for (int r = 0; r < 8; r++) {
                const int row = bm0 + wm * 32 + ms * 16 + r + 8 * lhi;
                const float val = acc[ms][ns][r] * scale + bv;
                int64_t idx;
                if constexpr (LAYOUT == 0) {
                    idx = (int64_t)z * c_bstride + (int64_t)row * ldc + col;
                } else if constexpr (LAYOUT == 1) {
                    idx = (int64_t)z * 1048576 + (int64_t)(col >> 7) * 131072
                        + (int64_t)row * 128 + (col & 127);
                } else {
                    idx = (int64_t)(z >> 3) * 1048576 + (int64_t)(z & 7) * 128
                        + (int64_t)row * 1024 + col;
                }
                Cp[idx] = (OT)val;
            }
        }
    }
}

// ---------------------------------------------------------------------------
// mask + softmax over rows of att [B*H*S, 1024]; f32 in place (d_out) and a
// f16 copy to workspace for the att@V WMMA GEMM.
// ---------------------------------------------------------------------------
__global__ __launch_bounds__(256) void softmax_mask(float* __restrict__ att,
                                                    _Float16* __restrict__ att16,
                                                    const unsigned char* __restrict__ mask) {
    __shared__ float red[256];
    const int t = threadIdx.x;
    const int64_t rowid = blockIdx.x;          // 0 .. B*H*S-1
    const int b = (int)(rowid >> 13);          // rowid / (H*S)
    float* rp = att + rowid * 1024;
    _Float16* hp = att16 + rowid * 1024;
    const unsigned char* mp = mask + (int64_t)b * 1024;

    float vloc[4];
    float mx = -3.4e38f;
    #pragma unroll
    for (int i = 0; i < 4; i++) {
        const int idx = t + i * 256;
        float x = rp[idx];
        if (mp[idx]) x = -1e9f;
        vloc[i] = x;
        mx = fmaxf(mx, x);
    }
    red[t] = mx; __syncthreads();
    for (int s = 128; s > 0; s >>= 1) {
        if (t < s) red[t] = fmaxf(red[t], red[t + s]);
        __syncthreads();
    }
    mx = red[0]; __syncthreads();

    float sum = 0.0f;
    #pragma unroll
    for (int i = 0; i < 4; i++) {
        vloc[i] = expf(vloc[i] - mx);
        sum += vloc[i];
    }
    red[t] = sum; __syncthreads();
    for (int s = 128; s > 0; s >>= 1) {
        if (t < s) red[t] += red[t + s];
        __syncthreads();
    }
    const float inv = 1.0f / red[0];
    #pragma unroll
    for (int i = 0; i < 4; i++) {
        const float y = vloc[i] * inv;
        rp[t + i * 256] = y;
        hp[t + i * 256] = (_Float16)y;
    }
}

// ---------------------------------------------------------------------------
extern "C" void kernel_launch(void* const* d_in, const int* in_sizes, int n_in,
                              void* d_out, int out_size, void* d_ws, size_t ws_size,
                              hipStream_t stream) {
    (void)in_sizes; (void)n_in; (void)out_size; (void)ws_size;
    const float* v     = (const float*)d_in[0];
    const float* key   = (const float*)d_in[1];
    const float* query = (const float*)d_in[2];
    const unsigned char* mask = (const unsigned char*)d_in[3];  // jax bool = 1 byte
    const float* Wq = (const float*)d_in[4];
    const float* bq = (const float*)d_in[5];
    const float* Wk = (const float*)d_in[6];
    const float* bk = (const float*)d_in[7];
    const float* Wv = (const float*)d_in[8];
    const float* bv = (const float*)d_in[9];
    const float* Wm = (const float*)d_in[10];
    const float* bm = (const float*)d_in[11];

    float* atted = (float*)d_out;                 // [16,1024,1024]
    float* att   = (float*)d_out + 16777216;      // [16,8,1024,1024]

    // Workspace layout (bytes). att16 [0,256MB) aliases qx/kx/vx/qh/kh, which
    // are all dead by the time softmax writes att16.
    char* ws = (char*)d_ws;
    _Float16* att16 = (_Float16*)(ws);                 // 256MB, written by softmax
    _Float16* qx    = (_Float16*)(ws);                 // f16 query  [B,S,RD] 32MB
    _Float16* kx    = (_Float16*)(ws + 33554432);      // f16 key    32MB
    _Float16* vx    = (_Float16*)(ws + 67108864);      // f16 v      32MB
    _Float16* qh    = (_Float16*)(ws + 100663296);     // q split-head 32MB
    _Float16* kh    = (_Float16*)(ws + 134217728);     // k split-head 32MB
    _Float16* vh    = (_Float16*)(ws + 268435456);     // v split-head 32MB (survives softmax)
    _Float16* Wq16  = (_Float16*)(ws + 301989888);     // 2MB each
    _Float16* Wk16  = (_Float16*)(ws + 304087040);
    _Float16* Wv16  = (_Float16*)(ws + 306184192);
    _Float16* Wm16  = (_Float16*)(ws + 308281344);
    _Float16* ap    = (_Float16*)(ws + 310378496);     // atted_pre [B,S,NH] 32MB

    // 0) convert inputs + weights to f16
    cvt_f32_f16_x4<<<16384, 256, 0, stream>>>(query, qx, 1 << 22);
    cvt_f32_f16_x4<<<16384, 256, 0, stream>>>(key,   kx, 1 << 22);
    cvt_f32_f16_x4<<<16384, 256, 0, stream>>>(v,     vx, 1 << 22);
    cvt_f32_f16_x4<<<1024, 256, 0, stream>>>(Wq, Wq16, 1 << 18);
    cvt_f32_f16_x4<<<1024, 256, 0, stream>>>(Wk, Wk16, 1 << 18);
    cvt_f32_f16_x4<<<1024, 256, 0, stream>>>(Wv, Wv16, 1 << 18);
    cvt_f32_f16_x4<<<1024, 256, 0, stream>>>(Wm, Wm16, 1 << 18);

    // 1) QKV projections: per-batch [1024,1024]x[1024,1024]+b -> split-head f16
    dim3 g1(8, 8, 16);
    gemm_wmma<false, 1, _Float16><<<g1, 256, 0, stream>>>(
        1024, qx, (int64_t)1 << 20, 1024, Wq16, 0, 1024, bq, 1.0f, qh, 0, 0);
    gemm_wmma<false, 1, _Float16><<<g1, 256, 0, stream>>>(
        1024, kx, (int64_t)1 << 20, 1024, Wk16, 0, 1024, bk, 1.0f, kh, 0, 0);
    gemm_wmma<false, 1, _Float16><<<g1, 256, 0, stream>>>(
        1024, vx, (int64_t)1 << 20, 1024, Wv16, 0, 1024, bv, 1.0f, vh, 0, 0);

    // 2) scores = (q @ k^T) / sqrt(dk), per (b,h) -> f32 att region of d_out
    dim3 g2(8, 8, 128);
    gemm_wmma<true, 0, float><<<g2, 256, 0, stream>>>(
        128, qh, 131072, 128, kh, 131072, 128, nullptr,
        0.08838834764831845f, att, 1048576, 1024);

    // 3) mask + softmax, in place + f16 copy
    softmax_mask<<<131072, 256, 0, stream>>>(att, att16, mask);

    // 4) atted_pre = att @ vv, merged-head f16 [B,S,NH]
    dim3 g4(8, 1, 128);
    gemm_wmma<false, 2, _Float16><<<g4, 256, 0, stream>>>(
        1024, att16, 1048576, 1024, vh, 131072, 128, nullptr, 1.0f, ap, 0, 0);

    // 5) atted = atted_pre @ Wm + bm, f32 out
    dim3 g5(128, 8, 1);
    gemm_wmma<false, 0, float><<<g5, 256, 0, stream>>>(
        1024, ap, 0, 1024, Wm16, 0, 1024, bm, 1.0f, atted, 0, 1024);
}
